// BidirectionalLSTM_18210661335501
// MI455X (gfx1250) — matrix operand order repaired and checked by reading, not compiled, not run on hardware
//
#include <hip/hip_runtime.h>
#include <cstdint>
#include <cstddef>

// ---------------- types for WMMA fragments ----------------
typedef __attribute__((ext_vector_type(16))) __bf16 v16bf;
typedef __attribute__((ext_vector_type(8)))  __bf16 v8bf;
typedef __attribute__((ext_vector_type(8)))  float  v8f;

#define S_LEN 256
#define B_SZ  64
#define I_DIM 512
#define H_DIM 512

// ---------------- workspace layout (bytes) ----------------
// xbf16  : 64*256*512 * 2              = 16,777,216
// WxT    : 8 * 512 * 512 * 2           =  4,194,304   (bf16, [dg][n][k], k = x-part)
// WhT    : 8 * 512 * 512 * 2           =  4,194,304   (bf16, [dg][n][k], k = h-part)
// bias   : 8 * 512 * 4                 =     16,384
// hbuf   : 2(par) * 2(dir) * 64*512 *2 =    262,144   (bf16 double buffer)
// c      : 2(dir) * 64*512 * 4         =    262,144
// P      : 2*256*4*64*512 * 4          = 268,435,456  (fp32 pre-activations)
#define OFF_XBF   ((size_t)0)
#define OFF_WXT   ((size_t)16777216)
#define OFF_WHT   ((size_t)20971520)
#define OFF_BIAS  ((size_t)25165824)
#define OFF_H     ((size_t)25182208)
#define OFF_C     ((size_t)25444352)
#define OFF_P     ((size_t)25706496)

struct GatePtrs { const float* W[8]; const float* b[8]; };

// fp32 -> bf16 bits, round-to-nearest-even
__device__ __forceinline__ unsigned short f2bf(float f) {
  union { float f; unsigned u; } x; x.f = f;
  unsigned u = x.u;
  if ((u & 0x7fffffffu) > 0x7f800000u) return (unsigned short)((u >> 16) | 0x0040u);
  return (unsigned short)((u + 0x7fffu + ((u >> 16) & 1u)) >> 16);
}

__device__ __forceinline__ float fast_sigmoid(float x) {
  return 1.0f / (1.0f + __expf(-x));
}
__device__ __forceinline__ float fast_tanh(float x) {
  // tanh(x) = 1 - 2/(exp(2x)+1); saturates correctly for large |x|
  return 1.0f - 2.0f / (__expf(2.0f * x) + 1.0f);
}

// A fragment (16x32 bf16, M x K), row-major source with row stride ldk elements.
// Contiguous k order used here: lanes 0-15: M=lane, K = 0..15; lanes 16-31: K = 16..31.
__device__ __forceinline__ v16bf load_a_frag(const unsigned short* base, int ldk, int lane) {
  int row = lane & 15;
  int kb  = (lane & 16) ? 16 : 0;
  const unsigned short* p = base + (size_t)row * ldk + kb;
  struct A2 { v8bf lo, hi; } t;
  t.lo = *(const v8bf*)(p);        // K = kb .. kb+7
  t.hi = *(const v8bf*)(p + 8);    // K = kb+8 .. kb+15
  return __builtin_bit_cast(v16bf, t);
}

// B fragment (32x16 bf16, K x N) from N-major (transposed) weights WT[n][k].
// Lanes 0-15: N=lane, K=0..15; lanes 16-31: K=16..31. One 32-byte load/lane.
__device__ __forceinline__ v16bf load_b_frag(const unsigned short* base, int ldk, int lane) {
  int n  = lane & 15;
  int ko = (lane & 16) ? 16 : 0;
  return *(const v16bf*)(base + (size_t)n * ldk + ko);
}

// ---------------- conversion kernels ----------------
__global__ void k_convert_x(const float* __restrict__ x,
                            unsigned short* __restrict__ xbf, int n) {
  int i = blockIdx.x * blockDim.x + threadIdx.x;
  int stride = gridDim.x * blockDim.x;
  for (; i < n; i += stride) xbf[i] = f2bf(x[i]);
}

__global__ void k_convert_w(GatePtrs gp,
                            unsigned short* __restrict__ WxT,
                            unsigned short* __restrict__ WhT) {
  // one thread per (dg, k, n): 8 * 1024 * 512 = 4,194,304 elements
  int idx = blockIdx.x * blockDim.x + threadIdx.x;
  int n  = idx & 511;
  int k  = (idx >> 9) & 1023;
  int dg = idx >> 19;
  unsigned short bf = f2bf(gp.W[dg][(size_t)k * H_DIM + n]);
  if (k < I_DIM) WxT[((size_t)dg * H_DIM + n) * I_DIM + k] = bf;
  else           WhT[((size_t)dg * H_DIM + n) * H_DIM + (k - I_DIM)] = bf;
}

__global__ void k_bias(GatePtrs gp, float* __restrict__ bias) {
  int idx = blockIdx.x * blockDim.x + threadIdx.x;   // 8*512 = 4096
  bias[idx] = gp.b[idx >> 9][idx & 511];
}

__global__ void k_init(unsigned short* __restrict__ h0, float* __restrict__ c) {
  int i = blockIdx.x * blockDim.x + threadIdx.x;     // 2*64*512 = 65536
  h0[i] = 0;
  c[i]  = 0.0f;
}

// ---------------- phase 1: P = x @ Wx + b  (gate-fused, A-fragment reuse) --------
__global__ __launch_bounds__(256)
void k_xgemm(const unsigned short* __restrict__ xbf,
             const unsigned short* __restrict__ WxT,
             const float* __restrict__ bias,
             float* __restrict__ P) {
  int lane = threadIdx.x & 31;
  int w = blockIdx.x * (blockDim.x >> 5) + (threadIdx.x >> 5);
  int ct = w & 31; w >>= 5;          // N tile (H/16)
  int st = w & 15; w >>= 4;          // S tile (16 timesteps)
  int b  = w & 63; w >>= 6;          // batch row
  int d  = w;                        // direction

  const unsigned short* abase = xbf + ((size_t)b * S_LEN + (size_t)st * 16) * I_DIM;

  v8f acc[4] = {v8f{}, v8f{}, v8f{}, v8f{}};
  #pragma unroll
  for (int kk = 0; kk < I_DIM; kk += 32) {
    v16bf a = load_a_frag(abase + kk, I_DIM, lane);   // shared by all 4 gates
    #pragma unroll
    for (int g = 0; g < 4; ++g) {
      const unsigned short* bbase =
          WxT + ((size_t)(d * 4 + g) * H_DIM + (size_t)ct * 16) * I_DIM + kk;
      v16bf bb = load_b_frag(bbase, I_DIM, lane);
      acc[g] = __builtin_amdgcn_wmma_f32_16x16x32_bf16(false, a, false, bb,
                                                       (short)0, acc[g], false, false);
    }
  }

  int n  = ct * 16 + (lane & 15);
  int mo = (lane & 16) ? 8 : 0;
  #pragma unroll
  for (int g = 0; g < 4; ++g) {
    float bv = bias[(d * 4 + g) * H_DIM + n];
    #pragma unroll
    for (int r = 0; r < 8; ++r) {
      int s = st * 16 + mo + r;
      size_t off = ((((size_t)d * S_LEN + s) * 4 + g) * B_SZ + b) * H_DIM + n;
      P[off] = acc[g][r] + bv;
    }
  }
}

// ---------------- phase 2: one recurrent step (both directions) ----------------
// Block = (direction, ct-pair); stages full h_prev[d] (64x512 bf16 = 64 KB) into
// LDS via async global->LDS copies, then 8 waves = 4 batch-tiles x 2 N-tiles.
__global__ __launch_bounds__(256)
void k_step(const unsigned short* __restrict__ hprev,   // [d][b][n] bf16
            unsigned short* __restrict__ hnext,         // [d][b][n] bf16
            const unsigned short* __restrict__ WhT,     // [dg][n][k] bf16
            const float* __restrict__ P,                // [d][s][g][b][n] fp32
            float* __restrict__ c,                      // [d][b][n] fp32
            float* __restrict__ out,                    // [b][s][2H] fp32
            int s) {
  __shared__ unsigned short sh[B_SZ * H_DIM];           // 64 KB h panel [b][k]

  int tid = threadIdx.x;
  int ctp = blockIdx.x & 15;        // ct pair
  int d   = blockIdx.x >> 4;        // direction

  // ---- async stage h_prev[d] into LDS (ASYNCcnt path) ----
  {
    unsigned lds_base = (unsigned)(uintptr_t)(&sh[0]);
    const char* gsrc = (const char*)(hprev + (size_t)d * B_SZ * H_DIM);
    #pragma unroll
    for (int i = 0; i < 16; ++i) {                      // 4096 x 16B chunks / 256 thr
      unsigned off = (unsigned)(tid + i * 256) * 16u;
      unsigned lo = lds_base + off;
      unsigned long long ga = (unsigned long long)(uintptr_t)(gsrc + off);
      asm volatile("global_load_async_to_lds_b128 %0, %1, off"
                   :: "v"(lo), "v"(ga) : "memory");
    }
    asm volatile("s_wait_asynccnt 0x0" ::: "memory");
  }
  __syncthreads();

  int lane = tid & 31;
  int wid  = tid >> 5;
  int bt   = wid & 3;               // batch tile (16 rows)
  int ct   = ctp * 2 + (wid >> 2);  // N tile

  const unsigned short* abase = &sh[0] + (size_t)(bt * 16) * H_DIM;

  v8f acc[4] = {v8f{}, v8f{}, v8f{}, v8f{}};
  #pragma unroll
  for (int kk = 0; kk < H_DIM; kk += 32) {
    v16bf a = load_a_frag(abase + kk, H_DIM, lane);     // LDS (ds_load) A fragment
    #pragma unroll
    for (int g = 0; g < 4; ++g) {
      const unsigned short* bbase =
          WhT + ((size_t)(d * 4 + g) * H_DIM + (size_t)ct * 16) * H_DIM + kk;
      v16bf bb = load_b_frag(bbase, H_DIM, lane);
      acc[g] = __builtin_amdgcn_wmma_f32_16x16x32_bf16(false, a, false, bb,
                                                       (short)0, acc[g], false, false);
    }
  }

  int n  = ct * 16 + (lane & 15);
  int mo = (lane & 16) ? 8 : 0;
  int sout = (d == 0) ? s : (S_LEN - 1 - s);
  const size_t gstride = (size_t)B_SZ * H_DIM;
  size_t pbase = (((size_t)d * S_LEN + s) * 4) * gstride;

  #pragma unroll
  for (int r = 0; r < 8; ++r) {
    int b = bt * 16 + mo + r;
    size_t bo = (size_t)b * H_DIM + n;
    float zi = acc[0][r] + P[pbase + 0 * gstride + bo];
    float zf = acc[1][r] + P[pbase + 1 * gstride + bo];
    float zg = acc[2][r] + P[pbase + 2 * gstride + bo];   // candidate (Wc)
    float zo = acc[3][r] + P[pbase + 3 * gstride + bo];
    float it = fast_sigmoid(zi);
    float ft = fast_sigmoid(zf);
    float ot = fast_sigmoid(zo);
    float gt = fast_tanh(zg);
    size_t co = (size_t)d * gstride + bo;
    float cv = ft * c[co] + it * gt;
    c[co] = cv;
    float hv = ot * fast_tanh(cv);
    out[((size_t)b * S_LEN + sout) * (2 * H_DIM) + (size_t)d * H_DIM + n] = hv;
    hnext[co] = f2bf(hv);
  }
}

// ---------------- host launcher ----------------
extern "C" void kernel_launch(void* const* d_in, const int* in_sizes, int n_in,
                              void* d_out, int out_size, void* d_ws, size_t ws_size,
                              hipStream_t stream) {
  (void)in_sizes; (void)n_in; (void)out_size; (void)ws_size;
  const float* x = (const float*)d_in[0];
  GatePtrs gp;
  for (int d = 0; d < 2; ++d)
    for (int g = 0; g < 4; ++g) {
      gp.W[d * 4 + g] = (const float*)d_in[1 + d * 8 + g * 2];
      gp.b[d * 4 + g] = (const float*)d_in[2 + d * 8 + g * 2];
    }

  char* ws = (char*)d_ws;
  unsigned short* xbf  = (unsigned short*)(ws + OFF_XBF);
  unsigned short* WxT  = (unsigned short*)(ws + OFF_WXT);
  unsigned short* WhT  = (unsigned short*)(ws + OFF_WHT);
  float*          bias = (float*)(ws + OFF_BIAS);
  unsigned short* hbuf = (unsigned short*)(ws + OFF_H);
  float*          cst  = (float*)(ws + OFF_C);
  float*          P    = (float*)(ws + OFF_P);
  float*          out  = (float*)d_out;

  const int NX = B_SZ * S_LEN * I_DIM;            // 8,388,608
  k_convert_x<<<4096, 256, 0, stream>>>(x, xbf, NX);
  k_convert_w<<<16384, 256, 0, stream>>>(gp, WxT, WhT);
  k_bias<<<16, 256, 0, stream>>>(gp, bias);
  k_init<<<256, 256, 0, stream>>>(hbuf, cst);     // zeros parity-0 h and c

  // phase 1: 2*64*16*32 = 65,536 wave-tiles (gate-fused), 8 waves/block
  k_xgemm<<<8192, 256, 0, stream>>>(xbf, WxT, bias, P);

  // phase 2: 256 sequential steps, double-buffered h
  const size_t hhalf = (size_t)2 * B_SZ * H_DIM;  // elements per parity
  for (int s = 0; s < S_LEN; ++s) {
    unsigned short* hp = hbuf + (size_t)(s & 1) * hhalf;
    unsigned short* hn = hbuf + (size_t)((s + 1) & 1) * hhalf;
    k_step<<<32, 256, 0, stream>>>(hp, hn, WhT, P, cst, out, s);
  }
}
